// PixelContrastLoss_46145128629053
// MI455X (gfx1250) — matrix operand order repaired and checked by reading, not compiled, not run on hardware
//
#include <hip/hip_runtime.h>

// ---------------------------------------------------------------------------
// SupCon loss on MI455X (gfx1250), wave32 + WMMA bf16 split-precision GEMM.
//   S = f f^T recomputed in two passes (flash-style stats, then log-term).
//   f stored as bf16 hi + bf16 residual -> 3x v_wmma_f32_16x16x32_bf16 per
//   32-wide k step gives ~f32 accuracy at bf16 matrix throughput.
//   Round 2: softmax stats kept PER LANE inside the j-sweep (no cross-lane
//   ops in the hot loop); 16-lane merge done once at the end. This removes
//   the ds_bpermute/v_cmp chains that were serializing the WMMA stream.
// ---------------------------------------------------------------------------

typedef __attribute__((ext_vector_type(16))) __bf16 v16bf;
typedef __attribute__((ext_vector_type(8)))  __bf16 v8bf;
typedef __attribute__((ext_vector_type(8)))  float  v8f;

#define DDIM   256
#define KSTEPS 8            // 256 / 32
#define MAXN   8192
#define INVT   (1.0f/0.07f)

__device__ __forceinline__ unsigned short f2bf(float x) {
  unsigned u = __float_as_uint(x);
  u += 0x7FFFu + ((u >> 16) & 1u);          // round-to-nearest-even
  return (unsigned short)(u >> 16);
}
__device__ __forceinline__ float bf2f(unsigned short b) {
  return __uint_as_float(((unsigned)b) << 16);
}

// Split f32 features into bf16 hi + bf16 residual (hi+lo trick).
__global__ void k_split(const float* __restrict__ f,
                        unsigned short* __restrict__ fhi,
                        unsigned short* __restrict__ flo, int total) {
  int i = blockIdx.x * blockDim.x + threadIdx.x;
  if (i >= total) return;
  float x = f[i];
  unsigned short h = f2bf(x);
  fhi[i] = h;
  flo[i] = f2bf(x - bf2f(h));
}

__device__ __forceinline__ v16bf pack16(v8bf a, v8bf b) {
  return __builtin_shufflevector(a, b, 0,1,2,3,4,5,6,7,8,9,10,11,12,13,14,15);
}

// A panel (16 rows x 256 k) for this wave, register resident.
// 16-bit A 16x32 layout: lane m = L&15; K block select by hi = L>>4.
// base = fhi + row*DDIM + 8*hi  (16B aligned)
__device__ __forceinline__ void load_A(const unsigned short* __restrict__ base,
                                       v16bf* A) {
#pragma unroll
  for (int k = 0; k < KSTEPS; ++k) {
    v8bf a0 = *(const v8bf*)(base + k * 32);       // K = 32k + 8*hi + 0..7
    v8bf a1 = *(const v8bf*)(base + k * 32 + 16);  // K = 32k + 16 + 8*hi + 0..7
    A[k] = pack16(a0, a1);
  }
}

// One 16x16 output tile: acc = Ahi*Bhi + Ahi*Blo + Alo*Bhi  (f32 accum).
// bh/bl = f{hi,lo} + (jb + (L&15))*DDIM + 8*(L>>4)   (B 32x16 layout).
__device__ __forceinline__ v8f tile_mma(const v16bf* __restrict__ Ah,
                                        const v16bf* __restrict__ Al,
                                        const unsigned short* __restrict__ bh,
                                        const unsigned short* __restrict__ bl) {
  v8f acc = {0.f,0.f,0.f,0.f,0.f,0.f,0.f,0.f};
#pragma unroll
  for (int k = 0; k < KSTEPS; ++k) {
    v8bf bh0 = *(const v8bf*)(bh + k * 32);
    v8bf bh1 = *(const v8bf*)(bh + k * 32 + 16);
    v8bf bl0 = *(const v8bf*)(bl + k * 32);
    v8bf bl1 = *(const v8bf*)(bl + k * 32 + 16);
    v16bf Bh = pack16(bh0, bh1);
    v16bf Bl = pack16(bl0, bl1);
    acc = __builtin_amdgcn_wmma_f32_16x16x32_bf16(false, Ah[k], false, Bh,
                                                  (short)0, acc, false, false);
    acc = __builtin_amdgcn_wmma_f32_16x16x32_bf16(false, Ah[k], false, Bl,
                                                  (short)0, acc, false, false);
    acc = __builtin_amdgcn_wmma_f32_16x16x32_bf16(false, Al[k], false, Bh,
                                                  (short)0, acc, false, false);
  }
  return acc;
}

// Pass A: per-row max, sum of exp over negatives, same-label count.
// One wave per 16-row strip. C layout: VGPR r, lanes 0-15 -> row r,
// lanes 16-31 -> row 8+r. Stats are per-lane online accumulators (each lane
// owns 512 columns); the 16-lane merge happens once after the sweep.
__global__ void __launch_bounds__(256) k_stats(
    const unsigned short* __restrict__ fhi, const unsigned short* __restrict__ flo,
    const int* __restrict__ labels,
    float* __restrict__ rmax, float* __restrict__ rnegs, float* __restrict__ rcnt,
    int Nrows) {
  __shared__ int slab[MAXN];
  for (int i = threadIdx.x; i < Nrows; i += blockDim.x) slab[i] = labels[i];
  __syncthreads();

  int lane  = threadIdx.x & 31;
  int wave  = threadIdx.x >> 5;
  int strip = blockIdx.x * (blockDim.x >> 5) + wave;
  int ib    = strip * 16;
  if (ib >= Nrows) return;

  int n  = lane & 15;
  int hi = lane >> 4;

  v16bf Ah[KSTEPS], Al[KSTEPS];
  {
    size_t aoff = (size_t)(ib + n) * DDIM + 8 * hi;
    load_A(fhi + aoff, Ah);
    load_A(flo + aoff, Al);
  }

  int labr[8];
#pragma unroll
  for (int r = 0; r < 8; ++r) labr[r] = slab[ib + r + 8 * hi];

  float m[8], ns[8], cnt[8];
#pragma unroll
  for (int r = 0; r < 8; ++r) { m[r] = -3.4e38f; ns[r] = 0.f; cnt[r] = 0.f; }

  for (int jb = 0; jb < Nrows; jb += 16) {
    size_t boff = (size_t)(jb + n) * DDIM + 8 * hi;
    v8f acc = tile_mma(Ah, Al, fhi + boff, flo + boff);
    int labj = slab[jb + n];
#pragma unroll
    for (int r = 0; r < 8; ++r) {
      float val  = acc[r] * INVT;
      bool  neg  = (labj != labr[r]);
      float mnew = fmaxf(m[r], val);
      float e    = neg ? __expf(val - mnew) : 0.f;
      ns[r]  = ns[r] * __expf(m[r] - mnew) + e;   // scale==1 when val<=m
      m[r]   = mnew;
      cnt[r] += neg ? 0.f : 1.f;
    }
  }

  // Merge the 16 per-lane accumulators of each half (log-sum-exp merge).
#pragma unroll
  for (int r = 0; r < 8; ++r) {
#pragma unroll
    for (int sh = 1; sh < 16; sh <<= 1) {
      float mo = __shfl_xor(m[r],   sh, 32);
      float no = __shfl_xor(ns[r],  sh, 32);
      float co = __shfl_xor(cnt[r], sh, 32);
      float mn = fmaxf(m[r], mo);
      ns[r] = ns[r] * __expf(m[r] - mn) + no * __expf(mo - mn);
      m[r]  = mn;
      cnt[r] += co;
    }
  }

  if (n == 0) {
#pragma unroll
    for (int r = 0; r < 8; ++r) {
      int row = ib + r + 8 * hi;
      rmax[row]  = m[r];
      rnegs[row] = ns[r];
      rcnt[row]  = cnt[r] - 1.f;   // exclude self
    }
  }
}

// Pass B: recompute tiles; accumulate sum over positives (excluding self) of
// (l - m) - log(exp(l - m) + neg_sum); fold into per-strip partial loss.
__global__ void __launch_bounds__(256) k_loss(
    const unsigned short* __restrict__ fhi, const unsigned short* __restrict__ flo,
    const int* __restrict__ labels,
    const float* __restrict__ rmax, const float* __restrict__ rnegs,
    const float* __restrict__ rcnt,
    float* __restrict__ partial, int Nrows) {
  __shared__ int slab[MAXN];
  for (int i = threadIdx.x; i < Nrows; i += blockDim.x) slab[i] = labels[i];
  __syncthreads();

  int lane  = threadIdx.x & 31;
  int wave  = threadIdx.x >> 5;
  int strip = blockIdx.x * (blockDim.x >> 5) + wave;
  int ib    = strip * 16;
  if (ib >= Nrows) return;

  int n  = lane & 15;
  int hi = lane >> 4;

  v16bf Ah[KSTEPS], Al[KSTEPS];
  {
    size_t aoff = (size_t)(ib + n) * DDIM + 8 * hi;
    load_A(fhi + aoff, Ah);
    load_A(flo + aoff, Al);
  }

  int labr[8];
  float m[8], ns[8], ct[8], pacc[8];
#pragma unroll
  for (int r = 0; r < 8; ++r) {
    int row = ib + r + 8 * hi;
    labr[r] = slab[row];
    m[r]    = rmax[row];
    ns[r]   = rnegs[row];
    ct[r]   = rcnt[row];
    pacc[r] = 0.f;
  }

  for (int jb = 0; jb < Nrows; jb += 16) {
    size_t boff = (size_t)(jb + n) * DDIM + 8 * hi;
    v8f acc = tile_mma(Ah, Al, fhi + boff, flo + boff);
    int labj = slab[jb + n];
    int jg   = jb + n;
#pragma unroll
    for (int r = 0; r < 8; ++r) {
      int row  = ib + r + 8 * hi;
      bool pos = (labj == labr[r]) && (jg != row);
      float l  = acc[r] * INVT - m[r];
      float t  = l - __logf(__expf(l) + ns[r]);
      pacc[r] += pos ? t : 0.f;
    }
  }

  float half = 0.f;
#pragma unroll
  for (int r = 0; r < 8; ++r) {
    float p = pacc[r];
#pragma unroll
    for (int sh = 1; sh < 16; sh <<= 1) p += __shfl_xor(p, sh, 32);
    half += -p / ct[r];                 // per-row loss, -(T/BT)=-1 * mean pos
  }
  half += __shfl_xor(half, 16, 32);     // combine the two lane-halves
  if (lane == 0) partial[strip] = half / (float)Nrows;
}

// Final deterministic reduction of per-strip partials into the scalar loss.
__global__ void k_reduce(const float* __restrict__ partial, float* __restrict__ out,
                         int nstrips) {
  __shared__ float s[256];
  float acc = 0.f;
  for (int i = threadIdx.x; i < nstrips; i += 256) acc += partial[i];
  s[threadIdx.x] = acc;
  __syncthreads();
  for (int w = 128; w > 0; w >>= 1) {
    if ((int)threadIdx.x < w) s[threadIdx.x] += s[threadIdx.x + w];
    __syncthreads();
  }
  if (threadIdx.x == 0) out[0] = s[0];
}

extern "C" void kernel_launch(void* const* d_in, const int* in_sizes, int n_in,
                              void* d_out, int out_size, void* d_ws, size_t ws_size,
                              hipStream_t stream) {
  const float* feats  = (const float*)d_in[0];
  const int*   labels = (const int*)d_in[1];
  int N     = in_sizes[1];           // 8192
  int total = in_sizes[0];           // N * 1 * 256

  char* ws = (char*)d_ws;
  size_t szHalf = (size_t)N * DDIM * sizeof(unsigned short);
  unsigned short* fhi = (unsigned short*)ws; ws += szHalf;
  unsigned short* flo = (unsigned short*)ws; ws += szHalf;
  float* rmax    = (float*)ws; ws += (size_t)N * sizeof(float);
  float* rnegs   = (float*)ws; ws += (size_t)N * sizeof(float);
  float* rcnt    = (float*)ws; ws += (size_t)N * sizeof(float);
  float* partial = (float*)ws;

  int nstrips = N / 16;
  dim3 blk(256);
  dim3 grd((nstrips + 7) / 8);       // 8 waves/block, one strip per wave

  k_split<<<(total + 255) / 256, 256, 0, stream>>>(feats, fhi, flo, total);
  k_stats<<<grd, blk, 0, stream>>>(fhi, flo, labels, rmax, rnegs, rcnt, N);
  k_loss <<<grd, blk, 0, stream>>>(fhi, flo, labels, rmax, rnegs, rcnt, partial, N);
  k_reduce<<<1, 256, 0, stream>>>(partial, (float*)d_out, nstrips);
}
